// pyfred_5875515261145
// MI455X (gfx1250) — matrix operand order repaired and checked by reading, not compiled, not run on hardware
//
#include <hip/hip_runtime.h>
#include <math.h>

// Problem constants (from reference)
#define NW    50257
#define RDIM  300
#define NA    10000
#define NHID  512
#define BATCH 32
#define TLEN  64
#define SRCL  64

#define KX     640              // 2*RDIM=600 padded to mult of 32
#define NWPAD  50304            // NW padded to mult of 64 (786*64)
#define NB     786              // logits blocks (64 cols each)
#define G3     1536             // 3*NHID
#define NEG_HUGE (-3.402823466e38f)

// ---------- WMMA types / helpers (CDNA5 gfx1250, wave32) ----------
typedef __attribute__((ext_vector_type(16))) __bf16         v16bf;
typedef __attribute__((ext_vector_type(8)))  float          v8f;
typedef __attribute__((ext_vector_type(8)))  unsigned short v8us;

union BFrag { v16bf b; v8us h[2]; };

__device__ __forceinline__ unsigned short f2bf(float f) {
  unsigned int u = __float_as_uint(f);
  unsigned int r = u + 0x7FFFu + ((u >> 16) & 1u);   // round-to-nearest-even
  return (unsigned short)(r >> 16);
}

// A fragment: 16xK tile of row-major bf16 matrix (rows rowBase..rowBase+15), k0 mult of 32.
// CDNA5 16-bit A layout: lane l (m=l&15, sel=l>>4) holds K = k0+sel*8..+7 and K = k0+16+sel*8..+7.
__device__ __forceinline__ v16bf load_frag_a(const unsigned short* A, int rowBase, int lda, int k0) {
  int l = threadIdx.x & 31;
  const unsigned short* p = A + (size_t)(rowBase + (l & 15)) * lda + k0 + ((l >> 4) << 3);
  BFrag f;
  f.h[0] = *(const v8us*)(p);
  f.h[1] = *(const v8us*)(p + 16);
  return f.b;
}

// B fragment for D = A x B with B[k][n] = W[n][k], W row-major (ldb = K stride).
// Lane l (n=l&15, sel=l>>4) holds K = k0+sel*8..+7 and K = k0+16+sel*8..+7 of row colBase+n.
__device__ __forceinline__ v16bf load_frag_b(const unsigned short* W, int colBase, int ldb, int k0) {
  int l = threadIdx.x & 31;
  const unsigned short* p = W + (size_t)(colBase + (l & 15)) * ldb + k0 + ((l >> 4) << 3);
  BFrag f;
  f.h[0] = *(const v8us*)(p);
  f.h[1] = *(const v8us*)(p + 16);
  return f.b;
}

__device__ __forceinline__ v8f wmma_bf16(v16bf a, v16bf b, v8f c) {
  return __builtin_amdgcn_wmma_f32_16x16x32_bf16(false, a, false, b, (short)0, c, false, false);
}

// ---------- one-time conversion kernels ----------
__global__ void k_conv_wm(const float* __restrict__ Wm, unsigned short* __restrict__ WmB) {
  const long long total = (long long)NWPAD * NHID;
  for (long long i = blockIdx.x * (long long)blockDim.x + threadIdx.x; i < total;
       i += (long long)gridDim.x * blockDim.x) {
    long long r = i / NHID;
    WmB[i] = (r < NW) ? f2bf(Wm[i]) : (unsigned short)0;
  }
}

__global__ void k_conv_wi(const float* __restrict__ Wi, unsigned short* __restrict__ WiB) {
  const int total = G3 * KX;
  for (int i = blockIdx.x * blockDim.x + threadIdx.x; i < total; i += gridDim.x * blockDim.x) {
    int r = i / KX, c = i % KX;
    WiB[i] = (c < 2 * RDIM) ? f2bf(Wi[r * (2 * RDIM) + c]) : (unsigned short)0;
  }
}

__global__ void k_conv_wh(const float* __restrict__ Wh, unsigned short* __restrict__ WhB) {
  const int total = G3 * NHID;
  for (int i = blockIdx.x * blockDim.x + threadIdx.x; i < total; i += gridDim.x * blockDim.x)
    WhB[i] = f2bf(Wh[i]);
}

// Build initial x = [A_emb[a] | W_emb[src[:,0]] | 0pad], h buffers from `hidden`.
__global__ void k_init(const float* __restrict__ A_emb, const float* __restrict__ W_emb,
                       const float* __restrict__ hidden, const int* __restrict__ a,
                       const int* __restrict__ src,
                       unsigned short* __restrict__ xB,
                       unsigned short* __restrict__ hB16, float* __restrict__ hF32) {
  for (int i = threadIdx.x; i < BATCH * KX; i += blockDim.x) {
    int b = i / KX, c = i % KX;
    float v = 0.f;
    if (c < RDIM)           v = A_emb[(size_t)a[b] * RDIM + c];
    else if (c < 2 * RDIM)  v = W_emb[(size_t)src[b * SRCL] * RDIM + (c - RDIM)];
    xB[i] = f2bf(v);
  }
  for (int i = threadIdx.x; i < BATCH * NHID; i += blockDim.x) {
    float v = hidden[i];
    hF32[i] = v;
    hB16[i] = f2bf(v);
  }
}

// ---------- per-step GRU: gi/gh GEMMs via WMMA + fused gating ----------
// grid: 32 blocks (16-col tiles of NHID); block: 64 threads = 2 waves (M tiles of 16).
__global__ void __launch_bounds__(64) k_gru(
    const unsigned short* __restrict__ xB, const unsigned short* __restrict__ WiB,
    const unsigned short* __restrict__ WhB,
    const float* __restrict__ bi, const float* __restrict__ bh,
    const unsigned short* __restrict__ hB16_in, const float* __restrict__ hF32_in,
    unsigned short* __restrict__ hB16_out, float* __restrict__ hF32_out) {
  const int ct = blockIdx.x;
  const int wave = threadIdx.x >> 5;
  const int l = threadIdx.x & 31;
  const int rowBase = wave * 16;

  v8f zero = {};
  v8f acc_i[3], acc_h[3];
#pragma unroll
  for (int g = 0; g < 3; ++g) { acc_i[g] = zero; acc_h[g] = zero; }

  // gi = x @ Wi^T  (K = 640)
#pragma unroll 4
  for (int kt = 0; kt < KX / 32; ++kt) {
    v16bf av = load_frag_a(xB, rowBase, KX, kt * 32);
#pragma unroll
    for (int g = 0; g < 3; ++g) {
      v16bf bv = load_frag_b(WiB, g * NHID + ct * 16, KX, kt * 32);
      acc_i[g] = wmma_bf16(av, bv, acc_i[g]);
    }
  }
  // gh = h @ Wh^T  (K = 512)
#pragma unroll 4
  for (int kt = 0; kt < NHID / 32; ++kt) {
    v16bf av = load_frag_a(hB16_in, rowBase, NHID, kt * 32);
#pragma unroll
    for (int g = 0; g < 3; ++g) {
      v16bf bv = load_frag_b(WhB, g * NHID + ct * 16, NHID, kt * 32);
      acc_h[g] = wmma_bf16(av, bv, acc_h[g]);
    }
  }

  // fused GRU gating; C/D layout: lane l holds col n=l&15, rows (l>>4)*8 + v.
  const int col = ct * 16 + (l & 15);
  const int rb  = rowBase + ((l >> 4) << 3);
  const float bir = bi[col],          bhr = bh[col];
  const float biz = bi[NHID + col],   bhz = bh[NHID + col];
  const float bin = bi[2*NHID + col], bhn = bh[2*NHID + col];
#pragma unroll
  for (int v = 0; v < 8; ++v) {
    int row = rb + v;
    float ir = acc_i[0][v] + bir, iz = acc_i[1][v] + biz, inn = acc_i[2][v] + bin;
    float hr = acc_h[0][v] + bhr, hz = acc_h[1][v] + bhz, hn  = acc_h[2][v] + bhn;
    float r = 1.f / (1.f + __expf(-(ir + hr)));
    float z = 1.f / (1.f + __expf(-(iz + hz)));
    float n = tanhf(inn + r * hn);
    float hp = hF32_in[row * NHID + col];
    float hnew = (1.f - z) * n + z * hp;
    hF32_out[row * NHID + col] = hnew;
    hB16_out[row * NHID + col] = f2bf(hnew);
  }
}

// ---------- logits GEMM (32 x NW x 512) + per-block argmax partials ----------
// grid: NB=786 blocks (64 cols each); block: 128 threads = 4 waves, one 16-col tile per wave.
// Skipped entirely when the teacher-forcing mask makes argmax irrelevant (and t != T-1).
__global__ void __launch_bounds__(128) k_logits(
    const unsigned short* __restrict__ hB16, const unsigned short* __restrict__ WmB,
    const float* __restrict__ bm, const int* __restrict__ tf_mask, int t,
    float* __restrict__ logits, float* __restrict__ pmax, int* __restrict__ pidx) {
  if (t != TLEN - 1 && tf_mask[t] > 0) return;   // uniform: argmax unused this step

  __shared__ float sval[BATCH][64 + 1];
  const int wave = threadIdx.x >> 5;
  const int l = threadIdx.x & 31;
  const int nb = (blockIdx.x * 4 + wave) * 16;

  v8f acc0 = {}, acc1 = {};
#pragma unroll 4
  for (int kt = 0; kt < NHID / 32; ++kt) {
    v16bf bv = load_frag_b(WmB, nb, NHID, kt * 32);
    acc0 = wmma_bf16(load_frag_a(hB16, 0,  NHID, kt * 32), bv, acc0);
    acc1 = wmma_bf16(load_frag_a(hB16, 16, NHID, kt * 32), bv, acc1);
  }

  const int col = nb + (l & 15);
  const float bias = (col < NW) ? bm[col] : 0.f;
  const int scol = wave * 16 + (l & 15);
#pragma unroll
  for (int half = 0; half < 2; ++half) {
    v8f acc = half ? acc1 : acc0;
    int rbase = half * 16 + ((l >> 4) << 3);
#pragma unroll
    for (int v = 0; v < 8; ++v) {
      int row = rbase + v;
      float val = acc[v] + bias;
      logits[(size_t)row * NWPAD + col] = val;
      sval[row][scol] = (col < NW) ? val : NEG_HUGE;
    }
  }
  __syncthreads();

  if (threadIdx.x < BATCH) {
    int row = threadIdx.x;
    float best = NEG_HUGE; int bidx = 0;
    int cbase = blockIdx.x * 64;
#pragma unroll 4
    for (int j = 0; j < 64; ++j) {
      float v = sval[row][j];
      if (v > best) { best = v; bidx = cbase + j; }   // strict > => first max (matches argmax)
    }
    pmax[row * NB + blockIdx.x] = best;
    pidx[row * NB + blockIdx.x] = bidx;
  }
}

// ---------- final argmax reduce, next-token select, refresh x embedding slice ----------
__global__ void __launch_bounds__(256) k_next(
    const float* __restrict__ pmax, const int* __restrict__ pidx,
    const int* __restrict__ tf_mask, const int* __restrict__ trg,
    const float* __restrict__ W_emb, int t,
    unsigned short* __restrict__ xB, float* __restrict__ rowmax) {
  __shared__ int nxt_s[BATCH];
  const int tf = tf_mask[t];
  if (threadIdx.x < BATCH) {
    int row = threadIdx.x;
    int top1 = 0;
    if (tf == 0 || t == TLEN - 1) {
      float best = NEG_HUGE; int bidx = 0;
      for (int j = 0; j < NB; ++j) {
        float v = pmax[row * NB + j];
        if (v > best) { best = v; bidx = pidx[row * NB + j]; }
      }
      top1 = bidx;
      if (t == TLEN - 1) rowmax[row] = best;
    }
    nxt_s[row] = (tf > 0) ? trg[row * TLEN + t] : top1;
  }
  __syncthreads();
  if (t == TLEN - 1) return;  // no further step
  for (int i = threadIdx.x; i < BATCH * RDIM; i += blockDim.x) {
    int b = i / RDIM, c = i % RDIM;
    xB[b * KX + RDIM + c] = f2bf(W_emb[(size_t)nxt_s[b] * RDIM + c]);
  }
}

// ---------- final log-softmax into outputs[:,0,:] ----------
__global__ void __launch_bounds__(256) k_softmax(
    const float* __restrict__ logits, const float* __restrict__ rowmax,
    float* __restrict__ out) {
  const int row = blockIdx.x;
  const float m = rowmax[row];
  __shared__ float sred[256];
  float s = 0.f;
  for (int c = threadIdx.x; c < NW; c += 256)
    s += __expf(logits[(size_t)row * NWPAD + c] - m);
  sred[threadIdx.x] = s;
  __syncthreads();
  for (int off = 128; off > 0; off >>= 1) {
    if (threadIdx.x < off) sred[threadIdx.x] += sred[threadIdx.x + off];
    __syncthreads();
  }
  const float lse = m + __logf(sred[0]);
  for (int c = threadIdx.x; c < NW; c += 256)
    out[(size_t)row * (TLEN * NW) + c] = logits[(size_t)row * NWPAD + c] - lse;
}

// ---------- host launch ----------
extern "C" void kernel_launch(void* const* d_in, const int* in_sizes, int n_in,
                              void* d_out, int out_size, void* d_ws, size_t ws_size,
                              hipStream_t stream) {
  const float* W_emb  = (const float*)d_in[0];
  const float* A_emb  = (const float*)d_in[1];
  const float* Wi     = (const float*)d_in[2];
  const float* Wh     = (const float*)d_in[3];
  const float* bi     = (const float*)d_in[4];
  const float* bh     = (const float*)d_in[5];
  const float* Wm     = (const float*)d_in[6];
  const float* bm     = (const float*)d_in[7];
  const float* hidden = (const float*)d_in[8];
  const int*   a      = (const int*)d_in[9];
  const int*   src    = (const int*)d_in[10];
  const int*   trg    = (const int*)d_in[11];
  const int*   tf     = (const int*)d_in[12];
  float* out = (float*)d_out;

  // Workspace carve-up (all 256B aligned)
  uint8_t* w = (uint8_t*)d_ws;
  size_t off = 0;
  auto carve = [&](size_t bytes) { void* p = w + off; off += (bytes + 255) & ~(size_t)255; return p; };
  unsigned short* WmB   = (unsigned short*)carve((size_t)NWPAD * NHID * 2);
  unsigned short* WiB   = (unsigned short*)carve((size_t)G3 * KX * 2);
  unsigned short* WhB   = (unsigned short*)carve((size_t)G3 * NHID * 2);
  unsigned short* xB    = (unsigned short*)carve((size_t)BATCH * KX * 2);
  unsigned short* hB16  = (unsigned short*)carve((size_t)2 * BATCH * NHID * 2);
  float*          hF32  = (float*)carve((size_t)2 * BATCH * NHID * 4);
  float*          logit = (float*)carve((size_t)BATCH * NWPAD * 4);
  float*          pmax  = (float*)carve((size_t)BATCH * NB * 4);
  int*            pidx  = (int*)carve((size_t)BATCH * NB * 4);
  float*          rwmax = (float*)carve((size_t)BATCH * 4);

  // Output is zero except [:,0,:]
  hipMemsetAsync(d_out, 0, (size_t)out_size * sizeof(float), stream);

  // One-time bf16 conversions (Wm bf16 = 51.5 MB -> resident in 192 MB L2 thereafter)
  k_conv_wm<<<2048, 256, 0, stream>>>(Wm, WmB);
  k_conv_wi<<<512, 256, 0, stream>>>(Wi, WiB);
  k_conv_wh<<<512, 256, 0, stream>>>(Wh, WhB);
  k_init<<<1, 256, 0, stream>>>(A_emb, W_emb, hidden, a, src, xB, hB16, hF32);

  const size_t hS16 = (size_t)BATCH * NHID;       // elements per h buffer
  for (int t = 0; t < TLEN; ++t) {
    int cur = t & 1, nxt = cur ^ 1;
    k_gru<<<NHID / 16, 64, 0, stream>>>(xB, WiB, WhB, bi, bh,
                                        hB16 + cur * hS16, hF32 + cur * hS16,
                                        hB16 + nxt * hS16, hF32 + nxt * hS16);
    k_logits<<<NB, 128, 0, stream>>>(hB16 + nxt * hS16, WmB, bm, tf, t,
                                     logit, pmax, pidx);
    k_next<<<1, 256, 0, stream>>>(pmax, pidx, tf, trg, W_emb, t, xB, rwmax);
  }
  k_softmax<<<BATCH, 256, 0, stream>>>(logit, rwmax, out);
}
